// MetaBayesLinearParallel_2594160247049
// MI455X (gfx1250) — matrix operand order repaired and checked by reading, not compiled, not run on hardware
//
#include <hip/hip_runtime.h>

typedef __bf16 bf16_t;
typedef __attribute__((ext_vector_type(16))) __bf16 v16bf;
typedef __attribute__((ext_vector_type(8)))  __bf16 v8bf;
typedef __attribute__((ext_vector_type(4)))  __bf16 v4bf;
typedef __attribute__((ext_vector_type(8)))  float  v8f;
typedef __attribute__((ext_vector_type(4)))  float  v4f;

namespace {
constexpr int kS   = 8;
constexpr int kB   = 256;
constexpr int kIN  = 2048;
constexpr int kOUT = 2048;

constexpr int TM   = 128;     // batch-rows per workgroup tile
constexpr int TN   = 128;     // out-cols  per workgroup tile
constexpr int TK   = 32;      // K-chunk staged in LDS (one WMMA depth)
constexpr int LDSS = TK + 8;  // padded row stride (bf16 elems) = 80B -> conflict-free b128 gathers
}

union AFrag { v16bf v; v8bf h[2]; };

// Next-chunk register staging: 4 float4 of x + 4 each of mu/sigma/eps.
struct Stage {
  v4f xr[4];
  v4f mr[4];
  v4f gr[4];
  v4f er[4];
};

__device__ __forceinline__ void stage_load(Stage& st,
    const float* __restrict__ xs, const float* __restrict__ ms,
    const float* __restrict__ gs, const float* __restrict__ es,
    int k0, int lrow8, int lcol8) {
  #pragma unroll
  for (int i = 0; i < 4; ++i) {
    const int r = lrow8 + 32 * i;
    const size_t off = (size_t)r * kIN + k0 + lcol8;
    st.xr[i] = *(const v4f*)&xs[off];
    st.mr[i] = *(const v4f*)&ms[off];
    st.gr[i] = *(const v4f*)&gs[off];
    st.er[i] = *(const v4f*)&es[off];
  }
}

__device__ __forceinline__ void stage_store(const Stage& st,
    bf16_t* __restrict__ A, bf16_t* __restrict__ B, int lrow8, int lcol8) {
  #pragma unroll
  for (int i = 0; i < 4; ++i) {
    const int r = lrow8 + 32 * i;
    *(v4bf*)&A[r * LDSS + lcol8] = __builtin_convertvector(st.xr[i], v4bf);
    const v4f w = st.mr[i] + st.gr[i] * st.er[i];   // fused W = mu + sigma*eps
    *(v4bf*)&B[r * LDSS + lcol8] = __builtin_convertvector(w, v4bf);
  }
}

// D = A(16x32 bf16) * B(32x16 bf16) + C(16x16 f32); one matrix per wave32.
__global__ __launch_bounds__(256) void noisy_linear_wmma(
    const float* __restrict__ x,    // (S, B, IN)
    const float* __restrict__ wmu,  // (OUT, IN)
    const float* __restrict__ wsg,  // (OUT, IN)
    const float* __restrict__ bmu,  // (OUT)
    const float* __restrict__ bsg,  // (OUT)
    const float* __restrict__ ew,   // (S, OUT, IN)
    const float* __restrict__ eb,   // (S, OUT)
    float* __restrict__ out)        // (S, B, OUT)
{
  __shared__ bf16_t lA[2][TM * LDSS]; // X tiles  (double buffered), 2 x 10 KB
  __shared__ bf16_t lB[2][TN * LDSS]; // W tiles  (double buffered), 2 x 10 KB

  const int s  = blockIdx.z;
  const int m0 = blockIdx.x * TM;
  const int n0 = blockIdx.y * TN;

  const int tid  = threadIdx.x;
  const int lane = tid & 31;
  const int wave = tid >> 5;         // 8 waves
  const int wm   = (wave >> 2) * 64; // wave sub-tile: 64 (M) x 32 (N)
  const int wn   = (wave & 3) * 32;

  // staging decomposition: 8 threads cover one 32-float row (4 x float4 each)
  const int lrow8 = tid >> 3;        // 0..31
  const int lcol8 = (tid & 7) * 4;   // f32 column within K-chunk

  const float* xs = x   + ((size_t)s * kB   + m0) * kIN;
  const float* es = ew  + ((size_t)s * kOUT + n0) * kIN;
  const float* ms = wmu + (size_t)n0 * kIN;
  const float* gs = wsg + (size_t)n0 * kIN;

  v8f acc[4][2] = {};
  Stage st;

  // prologue: stage chunk 0 into buffer 0
  stage_load(st, xs, ms, gs, es, 0, lrow8, lcol8);
  stage_store(st, lA[0], lB[0], lrow8, lcol8);
  __syncthreads();

  for (int k0 = 0; k0 < kIN; k0 += TK) {
    const int  cur  = (k0 / TK) & 1;
    const bool more = (k0 + TK) < kIN;

    // issue next chunk's global loads first: latency hides under the WMMAs
    if (more) stage_load(st, xs, ms, gs, es, k0 + TK, lrow8, lcol8);
    if (k0 + 2 * TK < kIN) {  // global_prefetch_b8, two chunks ahead
      __builtin_prefetch(&es[(size_t)lrow8 * kIN + k0 + 2 * TK + lcol8], 0, 0);
      __builtin_prefetch(&xs[(size_t)lrow8 * kIN + k0 + 2 * TK + lcol8], 0, 0);
    }

    // ---- compute on buffer `cur` ----
    {
      const bf16_t* A = lA[cur];
      const bf16_t* B = lB[cur];
      AFrag a[4], b[2];
      const int hrow = lane & 15;
      // A-layout (16-bit 16x32): lane<16 -> K {0..7,16..23}; lane>=16 -> +8
      const int akb = (lane >> 4) * 8;
      #pragma unroll
      for (int mf = 0; mf < 4; ++mf) {
        const bf16_t* p = &A[(wm + mf * 16 + hrow) * LDSS + akb];
        a[mf].h[0] = *(const v8bf*)p;         // ds_load_b128
        a[mf].h[1] = *(const v8bf*)(p + 16);  // ds_load_b128
      }
      // B-layout (32x16): lane<16 -> col=lane, K {0..7,16..23}; lane>=16 -> +8
      const int bkb = (lane >> 4) * 8;
      #pragma unroll
      for (int nf = 0; nf < 2; ++nf) {
        const bf16_t* p = &B[(wn + nf * 16 + hrow) * LDSS + bkb];
        b[nf].h[0] = *(const v8bf*)p;
        b[nf].h[1] = *(const v8bf*)(p + 16);
      }
      #pragma unroll
      for (int mf = 0; mf < 4; ++mf)
        #pragma unroll
        for (int nf = 0; nf < 2; ++nf)
          acc[mf][nf] = __builtin_amdgcn_wmma_f32_16x16x32_bf16(
              false, a[mf].v, false, b[nf].v, (short)0, acc[mf][nf],
              false, false);
    }

    // ---- write next chunk into the other buffer (no conflict with `cur` readers) ----
    if (more) stage_store(st, lA[cur ^ 1], lB[cur ^ 1], lrow8, lcol8);
    __syncthreads();
  }

  // ---- epilogue: bias add + store (C/D layout: VGPR r -> M=r+8*(lane>=16), N=lane&15) ----
  const int nlane = lane & 15;
  const int mhalf = (lane >> 4) * 8;
  #pragma unroll
  for (int nf = 0; nf < 2; ++nf) {
    const int n = n0 + wn + nf * 16 + nlane;
    const float bias = bmu[n] + bsg[n] * eb[(size_t)s * kOUT + n];
    #pragma unroll
    for (int mf = 0; mf < 4; ++mf) {
      #pragma unroll
      for (int r = 0; r < 8; ++r) {
        const int m = m0 + wm + mf * 16 + mhalf + r;
        out[((size_t)s * kB + m) * kOUT + n] = acc[mf][nf][r] + bias;
      }
    }
  }
}

extern "C" void kernel_launch(void* const* d_in, const int* in_sizes, int n_in,
                              void* d_out, int out_size, void* d_ws, size_t ws_size,
                              hipStream_t stream) {
  (void)in_sizes; (void)n_in; (void)out_size; (void)d_ws; (void)ws_size;
  const float* x   = (const float*)d_in[0];
  const float* wmu = (const float*)d_in[1];
  const float* wsg = (const float*)d_in[2];
  const float* bmu = (const float*)d_in[3];
  const float* bsg = (const float*)d_in[4];
  const float* ew  = (const float*)d_in[5];
  const float* eb  = (const float*)d_in[6];
  float* out = (float*)d_out;

  dim3 grid(kB / TM, kOUT / TN, kS);   // (2, 16, 8) = 256 workgroups
  noisy_linear_wmma<<<grid, 256, 0, stream>>>(x, wmu, wsg, bmu, bsg, ew, eb, out);
}
